// GNN_64518998720827
// MI455X (gfx1250) — matrix operand order repaired and checked by reading, not compiled, not run on hardware
//
#include <hip/hip_runtime.h>

typedef __attribute__((ext_vector_type(16))) _Float16 v16h;
typedef __attribute__((ext_vector_type(2)))  _Float16 v2h;
typedef __attribute__((ext_vector_type(8)))  float    v8f;

#define TILE_M 64
#define TILE_N 64
#define TILE_K 32
#define LDSS   40    // LDS row stride in halves: 80B rows -> every 16B chunk aligned
#define NCOL   256   // every GEMM in this model has N == D == 256 (compile-time)

#ifndef USE_ASYNC_LDS
#define USE_ASYNC_LDS 1
#endif

// ---------------------------------------------------------------------------
// Tiled f16 WMMA GEMM: C[M,NCOL] = A[M,K] (or A[K,M]^T) * B[K,NCOL], f32 acc.
// 128 threads = 4 waves, each wave owns a 32x32 quadrant (2x2 wmma 16x16x32).
// Double-buffered LDS; NN A-tiles stream via global_load_async_to_lds_b128
// (ASYNCcnt), transposing paths use a register-staged pipeline.
// ---------------------------------------------------------------------------
template<bool TRANSA, bool OUTF16>
__global__ __launch_bounds__(128)
void wmma_gemm_kernel(const _Float16* __restrict__ A,
                      const _Float16* __restrict__ Bm,
                      float* __restrict__ Cf, _Float16* __restrict__ Ch,
                      int Mg, int Kg,
                      long long sA, long long sB, long long sC)
{
    __shared__ __align__(16) _Float16 As[2][TILE_M][LDSS];  // [m][k]
    __shared__ __align__(16) _Float16 Bs[2][TILE_N][LDSS];  // [n][k]

    const int tid  = threadIdx.x;
    const int wave = tid >> 5;
    const int lane = tid & 31;
    const int hl   = lane >> 4;
    const int l16  = lane & 15;
    const int m0   = blockIdx.y * TILE_M;
    const int n0   = blockIdx.x * TILE_N;
    const long long bb = blockIdx.z;
    const _Float16* Ab = A  + bb * sA;
    const _Float16* Bb = Bm + bb * sB;
    const int mW  = (wave >> 1) * 32;
    const int nW  = (wave & 1) * 32;
    const int lda = TRANSA ? Mg : Kg;
    const int nk  = Kg / TILE_K;

    v8f acc[2][2] = {};
    float4 regA[2], regB[2];

    // ---- tile movers -----------------------------------------------------
    auto loadB_g = [&](int kt) {
        #pragma unroll
        for (int it = 0; it < 2; ++it) {
            int v = tid + it * 128;
            int r = v >> 3, c = (v & 7) * 8;       // r = k row, c = n col
            regB[it] = *reinterpret_cast<const float4*>(
                Bb + (long long)(kt * TILE_K + r) * NCOL + n0 + c);
        }
    };
    auto storeB_s = [&](int buf) {                 // transpose into Bs[n][k]
        #pragma unroll
        for (int it = 0; it < 2; ++it) {
            int v = tid + it * 128;
            int r = v >> 3, c = (v & 7) * 8;
            const _Float16* h = reinterpret_cast<const _Float16*>(&regB[it]);
            #pragma unroll
            for (int j = 0; j < 8; ++j) Bs[buf][c + j][r] = h[j];
        }
    };
    auto loadA_reg = [&](int kt) {                 // TRANSA: A stored [K,M]
        #pragma unroll
        for (int it = 0; it < 2; ++it) {
            int v = tid + it * 128;
            int r = v >> 3, c = (v & 7) * 8;       // r = k row, c = m col
            regA[it] = *reinterpret_cast<const float4*>(
                Ab + (long long)(kt * TILE_K + r) * lda + m0 + c);
        }
    };
    auto storeA_reg = [&](int buf) {               // transpose into As[m][k]
        #pragma unroll
        for (int it = 0; it < 2; ++it) {
            int v = tid + it * 128;
            int r = v >> 3, c = (v & 7) * 8;
            const _Float16* h = reinterpret_cast<const _Float16*>(&regA[it]);
            #pragma unroll
            for (int j = 0; j < 8; ++j) As[buf][c + j][r] = h[j];
        }
    };
    auto issueA_async = [&](int kt, int buf) {     // !TRANSA: contiguous 16B chunks
        #pragma unroll
        for (int it = 0; it < 2; ++it) {
            int v = tid + it * 128;
            int r = v >> 2, c = (v & 3) * 8;       // r = m row, c = k col
#if USE_ASYNC_LDS
            // LDS aperture: low 32 bits of the flat shared address == LDS offset
            unsigned lds_off = (unsigned)(uintptr_t)&As[buf][r][c];
            const void* g = Ab + (long long)(m0 + r) * lda + kt * TILE_K + c;
            asm volatile("global_load_async_to_lds_b128 %0, %1, off"
                         :: "v"(lds_off), "v"(g) : "memory");
#else
            float4 d = *reinterpret_cast<const float4*>(
                Ab + (long long)(m0 + r) * lda + kt * TILE_K + c);
            const _Float16* h = reinterpret_cast<const _Float16*>(&d);
            #pragma unroll
            for (int j = 0; j < 8; j += 2)
                *reinterpret_cast<v2h*>(&As[buf][r][c + j]) =
                    *reinterpret_cast<const v2h*>(&h[j]);
#endif
        }
    };
    auto wait_async = [&]() {
#if USE_ASYNC_LDS
        asm volatile("s_wait_asynccnt 0x0" ::: "memory");
#endif
    };

    // ---- compute: build fragments per ISA 7.12.2 wave32 layouts ----------
    auto compute = [&](int buf) {
        v16h afr[2], bfr[2];
        #pragma unroll
        for (int i = 0; i < 2; ++i) {
            const _Float16* arow = &As[buf][mW + i * 16 + l16][0];
            #pragma unroll
            for (int v = 0; v < 8; ++v) {
                // A 16x32 f16: v<4 -> K = 8*half + 2v ; v>=4 -> 16 + 8*half + 2(v-4)
                int kb = (v < 4) ? (8 * hl + 2 * v) : (16 + 8 * hl + 2 * (v - 4));
                v2h p = *reinterpret_cast<const v2h*>(&arow[kb]);
                afr[i][2 * v] = p[0]; afr[i][2 * v + 1] = p[1];
            }
        }
        #pragma unroll
        for (int j = 0; j < 2; ++j) {
            const _Float16* brow = &Bs[buf][nW + j * 16 + l16][0];
            #pragma unroll
            for (int v = 0; v < 8; ++v) {
                // B 32x16 f16: v<4 -> K = 8*half + 2v ; v>=4 -> 16 + 8*half + 2(v-4)
                int kb = (v < 4) ? (8 * hl + 2 * v) : (16 + 8 * hl + 2 * (v - 4));
                v2h p = *reinterpret_cast<const v2h*>(&brow[kb]);
                bfr[j][2 * v] = p[0]; bfr[j][2 * v + 1] = p[1];
            }
        }
        #pragma unroll
        for (int i = 0; i < 2; ++i)
            #pragma unroll
            for (int j = 0; j < 2; ++j)
                acc[i][j] = __builtin_amdgcn_wmma_f32_16x16x32_f16(
                    false, afr[i], false, bfr[j],
                    (short)0, acc[i][j], false, false);
    };

    // ---- software pipeline: prefetch tile kt+1 while computing kt --------
    if (!TRANSA) issueA_async(0, 0);
    else { loadA_reg(0); storeA_reg(0); }
    loadB_g(0);
    storeB_s(0);
    wait_async();
    __syncthreads();

    int cur = 0;
    for (int kt = 0; kt < nk; ++kt) {
        const bool have = (kt + 1) < nk;
        if (have) {
            if (!TRANSA) issueA_async(kt + 1, cur ^ 1);
            else loadA_reg(kt + 1);
            loadB_g(kt + 1);
        }
        if (kt + 2 < nk) {  // warm L2 for the K+2 B tile (global_prefetch_b8)
            int r = tid >> 3, c = (tid & 7) * 8;
            __builtin_prefetch(Bb + (long long)((kt + 2) * TILE_K + r) * NCOL + n0 + c, 0, 1);
        }
        compute(cur);
        if (have) {
            if (TRANSA) storeA_reg(cur ^ 1);
            storeB_s(cur ^ 1);
            wait_async();
        }
        __syncthreads();
        cur ^= 1;
    }

    // ---- writeback: C/D layout VGPR e -> M = e + 8*half, N = lane%16 -----
    const int colBase = n0 + nW + l16;
    #pragma unroll
    for (int i = 0; i < 2; ++i) {
        const int rowBase = m0 + mW + i * 16 + 8 * hl;
        #pragma unroll
        for (int j = 0; j < 2; ++j) {
            const long long base = bb * sC + (long long)rowBase * NCOL + colBase + j * 16;
            #pragma unroll
            for (int e = 0; e < 8; ++e) {
                if (OUTF16) Ch[base + e * NCOL] = (_Float16)acc[i][j][e];
                else        Cf[base + e * NCOL] = acc[i][j][e];
            }
        }
    }
}

// ---------------------------------------------------------------------------
// Elementwise / reduction helpers
// ---------------------------------------------------------------------------
__device__ __forceinline__ float sigmoidf_(float x) { return 1.0f / (1.0f + expf(-x)); }

__global__ void k_f32_to_f16(const float* __restrict__ s, _Float16* __restrict__ d, long long n) {
    long long i = blockIdx.x * (long long)blockDim.x + threadIdx.x;
    if (i < n) d[i] = (_Float16)s[i];
}

__global__ void k_transpose_f16(const float* __restrict__ W, _Float16* __restrict__ WT,
                                int R, int C) {
    long long i = blockIdx.x * (long long)blockDim.x + threadIdx.x;
    if (i < (long long)R * C) {
        int r = (int)(i / C), c = (int)(i % C);
        WT[(long long)c * R + r] = (_Float16)W[i];
    }
}

__global__ void k_rowsum_norm(const float* __restrict__ M, float* __restrict__ norm, int L) {
    __shared__ float red[256];
    long long row = blockIdx.x;
    const float* p = M + row * (long long)L;
    float s = 0.f;
    for (int i = threadIdx.x; i < L; i += 256) s += p[i];
    red[threadIdx.x] = s; __syncthreads();
    for (int o = 128; o > 0; o >>= 1) {
        if (threadIdx.x < o) red[threadIdx.x] += red[threadIdx.x + o];
        __syncthreads();
    }
    if (threadIdx.x == 0) norm[row] = red[0] + 1.0f;
}

__global__ void k_colsum_norm(const float* __restrict__ n2e, float* __restrict__ norm,
                              int Bt, int Nn, int Ee) {
    long long i = blockIdx.x * (long long)blockDim.x + threadIdx.x;
    if (i >= (long long)Bt * Nn) return;
    long long b = i / Nn; int n = (int)(i % Nn);
    const float* p = n2e + b * (long long)Ee * Nn + n;
    float s = 0.f;
    for (int e = 0; e < Ee; ++e) s += p[(long long)e * Nn];
    norm[i] = s + 1.0f;
}

__global__ void k_fuse_epilogue(const float* __restrict__ tmp, const float* __restrict__ ev,
                                const float* __restrict__ bias, _Float16* __restrict__ out16,
                                long long total, int D) {
    long long i = blockIdx.x * (long long)blockDim.x + threadIdx.x;
    if (i < total) {
        float v = tmp[i] + ev[i] + bias[i % D];
        out16[i] = (_Float16)fmaxf(v, 0.0f);
    }
}

__global__ void k_build_inout(const float* __restrict__ agg_out, const float* __restrict__ agg_in,
                              const float* __restrict__ cur,
                              const float* __restrict__ norm_out, const float* __restrict__ norm_in,
                              float* __restrict__ out_f, float* __restrict__ in_f,
                              _Float16* __restrict__ feats, long long total, int D) {
    long long i = blockIdx.x * (long long)blockDim.x + threadIdx.x;
    if (i >= total) return;
    long long bn = i / D; int d = (int)(i % D);
    float c  = cur[i];
    float ov = (agg_out[i] + c) / norm_out[bn];
    float iv = (agg_in[i]  + c) / norm_in[bn];
    out_f[i] = ov; in_f[i] = iv;
    long long base = bn * (long long)(4 * D) + d;
    feats[base]         = (_Float16)iv;
    feats[base + D]     = (_Float16)ov;
    feats[base + 2 * D] = (_Float16)(iv * ov);
    feats[base + 3 * D] = (_Float16)(iv - ov);
}

__global__ void k_gated_epilogue(const float* __restrict__ ztmp, const float* __restrict__ bz,
                                 const float* __restrict__ in_f, const float* __restrict__ out_f,
                                 const float* __restrict__ cur, float* __restrict__ new_f,
                                 _Float16* __restrict__ cn, long long total, int D) {
    long long i = blockIdx.x * (long long)blockDim.x + threadIdx.x;
    if (i >= total) return;
    long long bn = i / D; int d = (int)(i % D);
    float z = sigmoidf_(ztmp[i] + bz[d]);
    float nv = (1.0f - z) * in_f[i] + z * out_f[i];
    new_f[i] = nv;
    long long base = bn * (long long)(2 * D) + d;
    cn[base]     = (_Float16)cur[i];
    cn[base + D] = (_Float16)nv;
}

__global__ void k_gru_mid(const float* __restrict__ utmp, const float* __restrict__ rtmp,
                          const float* __restrict__ cur, const float* __restrict__ new_f,
                          float* __restrict__ u_sig, _Float16* __restrict__ rn,
                          long long total, int D) {
    long long i = blockIdx.x * (long long)blockDim.x + threadIdx.x;
    if (i >= total) return;
    long long bn = i / D; int d = (int)(i % D);
    float u = sigmoidf_(utmp[i]);
    float r = sigmoidf_(rtmp[i]);
    u_sig[i] = u;
    long long base = bn * (long long)(2 * D) + d;
    rn[base]     = (_Float16)(r * cur[i]);
    rn[base + D] = (_Float16)new_f[i];
}

__global__ void k_gru_final(const float* __restrict__ mtmp, const float* __restrict__ u_sig,
                            float* __restrict__ cur, _Float16* __restrict__ node16,
                            long long total) {
    long long i = blockIdx.x * (long long)blockDim.x + threadIdx.x;
    if (i >= total) return;
    float m = tanhf(mtmp[i]);
    float u = u_sig[i];
    float v = (1.0f - u) * cur[i] + u * m;
    cur[i] = v;
    node16[i] = (_Float16)v;
}

__global__ void k_max_over_nodes(const float* __restrict__ t, float* __restrict__ out,
                                 int Bt, int Nn, int D) {
    long long i = blockIdx.x * (long long)blockDim.x + threadIdx.x;
    if (i >= (long long)Bt * D) return;
    long long b = i / D; int d = (int)(i % D);
    const float* p = t + b * (long long)Nn * D + d;
    float m = -3.402823466e38f;
    for (int n = 0; n < Nn; ++n) m = fmaxf(m, p[(long long)n * D]);
    out[i] = m;
}

__global__ void k_write_transposed(const float* __restrict__ cur, float* __restrict__ out,
                                   int Bt, int Nn, int D) {
    long long i = blockIdx.x * (long long)blockDim.x + threadIdx.x;
    if (i >= (long long)Bt * Nn * D) return;
    long long b = i / ((long long)Nn * D);
    long long rem = i % ((long long)Nn * D);
    long long n = rem / D;
    int d = (int)(rem % D);
    out[(n * Bt + b) * D + d] = cur[i];
}

// ---------------------------------------------------------------------------
static inline void launch_gemm(hipStream_t stream, bool transA, bool outF16,
                               const _Float16* A, const _Float16* Bm, void* C,
                               int M, int K,
                               long long sA, long long sB, long long sC, int batches)
{
    dim3 grid(NCOL / TILE_N, M / TILE_M, batches);
    dim3 block(128);
    if (!transA && !outF16)
        wmma_gemm_kernel<false, false><<<grid, block, 0, stream>>>(A, Bm, (float*)C, nullptr, M, K, sA, sB, sC);
    else if (!transA && outF16)
        wmma_gemm_kernel<false, true ><<<grid, block, 0, stream>>>(A, Bm, nullptr, (_Float16*)C, M, K, sA, sB, sC);
    else if (transA && !outF16)
        wmma_gemm_kernel<true,  false><<<grid, block, 0, stream>>>(A, Bm, (float*)C, nullptr, M, K, sA, sB, sC);
    else
        wmma_gemm_kernel<true,  true ><<<grid, block, 0, stream>>>(A, Bm, nullptr, (_Float16*)C, M, K, sA, sB, sC);
}

extern "C" void kernel_launch(void* const* d_in, const int* in_sizes, int n_in,
                              void* d_out, int out_size, void* d_ws, size_t ws_size,
                              hipStream_t stream)
{
    const int B = 8, Nn = 2048, E = 4096, D = 256, HOPS = 3;

    const float* node_vec  = (const float*)d_in[0];
    const float* edge_vec  = (const float*)d_in[1];
    const float* node2edge = (const float*)d_in[2];   // [B,E,Nn]
    const float* edge2node = (const float*)d_in[3];   // [B,Nn,E]
    /* d_in[4] = node_mask (unused by reference) */
    const float* W_fuse = (const float*)d_in[5];
    const float* b_fuse = (const float*)d_in[6];
    const float* W_z    = (const float*)d_in[7];
    const float* b_z    = (const float*)d_in[8];
    const float* W_u    = (const float*)d_in[9];
    const float* W_r    = (const float*)d_in[10];
    const float* W_m    = (const float*)d_in[11];
    const float* W_max  = (const float*)d_in[12];

    char* ws = (char*)d_ws;
    size_t off = 0;
    auto alloc = [&](size_t bytes) -> void* {
        void* p = ws + off;
        off = (off + bytes + 255) & ~(size_t)255;
        return p;
    };
    const long long nAdj = (long long)B * E * Nn;
    const long long nED  = (long long)B * E * D;
    const long long nND  = (long long)B * Nn * D;

    _Float16* n2e16   = (_Float16*)alloc(nAdj * 2);
    _Float16* e2n16   = (_Float16*)alloc(nAdj * 2);
    _Float16* node16  = (_Float16*)alloc(nND * 2);
    _Float16* edge16  = (_Float16*)alloc(nED * 2);
    _Float16* wfuseT  = (_Float16*)alloc((size_t)2 * D * D * 2);
    _Float16* wzT     = (_Float16*)alloc((size_t)4 * D * D * 2);
    _Float16* wuT     = (_Float16*)alloc((size_t)2 * D * D * 2);
    _Float16* wrT     = (_Float16*)alloc((size_t)2 * D * D * 2);
    _Float16* wmT     = (_Float16*)alloc((size_t)2 * D * D * 2);
    _Float16* wmaxT   = (_Float16*)alloc((size_t)D * D * 2);
    float*    ev_part = (float*)alloc(nED * 4);
    float*    norm_o  = (float*)alloc((size_t)B * Nn * 4);
    float*    norm_i  = (float*)alloc((size_t)B * Nn * 4);
    _Float16* ge_o16  = (_Float16*)alloc(nED * 2);
    _Float16* ge_i16  = (_Float16*)alloc(nED * 2);
    _Float16* eh_o16  = (_Float16*)alloc(nED * 2);
    _Float16* eh_i16  = (_Float16*)alloc(nED * 2);
    float*    tmp1    = (float*)alloc(nED * 4);
    float*    tmp2    = (float*)alloc(nND * 4);
    float*    agg_o   = (float*)alloc(nND * 4);
    float*    agg_i   = (float*)alloc(nND * 4);
    float*    out_f   = (float*)alloc(nND * 4);
    float*    in_f    = (float*)alloc(nND * 4);
    _Float16* feats16 = (_Float16*)alloc(nND * 4 * 2);
    _Float16* cn16    = (_Float16*)alloc(nND * 2 * 2);
    float*    new_f   = (float*)alloc(nND * 4);
    float*    u_sig   = (float*)alloc(nND * 4);
    _Float16* rn16    = (_Float16*)alloc(nND * 2 * 2);
    float*    node_cur= (float*)alloc(nND * 4);
    (void)ws_size; (void)in_sizes; (void)n_in; (void)out_size;

    auto GS = [](long long n) { return dim3((unsigned)((n + 255) / 256)); };

    // -------- hop-invariant prologue --------
    k_f32_to_f16<<<GS(nAdj), 256, 0, stream>>>(node2edge, n2e16, nAdj);
    k_f32_to_f16<<<GS(nAdj), 256, 0, stream>>>(edge2node, e2n16, nAdj);
    k_f32_to_f16<<<GS(nED),  256, 0, stream>>>(edge_vec, edge16, nED);
    k_f32_to_f16<<<GS(nND),  256, 0, stream>>>(node_vec, node16, nND);
    hipMemcpyAsync(node_cur, node_vec, (size_t)nND * 4, hipMemcpyDeviceToDevice, stream);

    k_transpose_f16<<<GS((long long)D * 2 * D), 256, 0, stream>>>(W_fuse, wfuseT, D, 2 * D);
    k_transpose_f16<<<GS((long long)D * 4 * D), 256, 0, stream>>>(W_z,    wzT,    D, 4 * D);
    k_transpose_f16<<<GS((long long)D * 2 * D), 256, 0, stream>>>(W_u,    wuT,    D, 2 * D);
    k_transpose_f16<<<GS((long long)D * 2 * D), 256, 0, stream>>>(W_r,    wrT,    D, 2 * D);
    k_transpose_f16<<<GS((long long)D * 2 * D), 256, 0, stream>>>(W_m,    wmT,    D, 2 * D);
    k_transpose_f16<<<GS((long long)D * D),     256, 0, stream>>>(W_max,  wmaxT,  D, D);

    k_rowsum_norm<<<dim3(B * Nn), 256, 0, stream>>>(edge2node, norm_o, E);
    k_colsum_norm<<<GS((long long)B * Nn), 256, 0, stream>>>(node2edge, norm_i, B, Nn, E);

    // ev_part = edge_vec16 @ W_fuse^T[D:2D, :]  (constant across hops)
    launch_gemm(stream, false, false, edge16, wfuseT + (size_t)D * D, ev_part,
                B * E, D, 0, 0, 0, 1);

    // -------- hop loop --------
    for (int h = 0; h < HOPS; ++h) {
        launch_gemm(stream, false, true, n2e16, node16, ge_o16,
                    E, Nn, (long long)E * Nn, (long long)Nn * D, (long long)E * D, B);
        launch_gemm(stream, true, true, e2n16, node16, ge_i16,
                    E, Nn, (long long)Nn * E, (long long)Nn * D, (long long)E * D, B);

        launch_gemm(stream, false, false, ge_o16, wfuseT, tmp1, B * E, D, 0, 0, 0, 1);
        k_fuse_epilogue<<<GS(nED), 256, 0, stream>>>(tmp1, ev_part, b_fuse, eh_o16, nED, D);
        launch_gemm(stream, false, false, ge_i16, wfuseT, tmp1, B * E, D, 0, 0, 0, 1);
        k_fuse_epilogue<<<GS(nED), 256, 0, stream>>>(tmp1, ev_part, b_fuse, eh_i16, nED, D);

        launch_gemm(stream, false, false, e2n16, eh_o16, agg_o,
                    Nn, E, (long long)Nn * E, (long long)E * D, (long long)Nn * D, B);
        launch_gemm(stream, true, false, n2e16, eh_i16, agg_i,
                    Nn, E, (long long)E * Nn, (long long)E * D, (long long)Nn * D, B);

        k_build_inout<<<GS(nND), 256, 0, stream>>>(agg_o, agg_i, node_cur,
                                                   norm_o, norm_i, out_f, in_f,
                                                   feats16, nND, D);
        launch_gemm(stream, false, false, feats16, wzT, tmp2, B * Nn, 4 * D, 0, 0, 0, 1);
        k_gated_epilogue<<<GS(nND), 256, 0, stream>>>(tmp2, b_z, in_f, out_f, node_cur,
                                                      new_f, cn16, nND, D);
        launch_gemm(stream, false, false, cn16, wuT, tmp1, B * Nn, 2 * D, 0, 0, 0, 1);
        launch_gemm(stream, false, false, cn16, wrT, tmp2, B * Nn, 2 * D, 0, 0, 0, 1);
        k_gru_mid<<<GS(nND), 256, 0, stream>>>(tmp1, tmp2, node_cur, new_f, u_sig, rn16, nND, D);
        launch_gemm(stream, false, false, rn16, wmT, tmp1, B * Nn, 2 * D, 0, 0, 0, 1);
        k_gru_final<<<GS(nND), 256, 0, stream>>>(tmp1, u_sig, node_cur, node16, nND);
    }

    // -------- epilogue: graph pooling + output layout --------
    launch_gemm(stream, false, false, node16, wmaxT, tmp1, B * Nn, D, 0, 0, 0, 1);
    float* out_nodes = (float*)d_out;                         // [Nn,B,D]
    float* out_gemb  = (float*)d_out + (long long)Nn * B * D; // [1,B,D]
    k_max_over_nodes<<<GS((long long)B * D), 256, 0, stream>>>(tmp1, out_gemb, B, Nn, D);
    k_write_transposed<<<GS(nND), 256, 0, stream>>>(node_cur, out_nodes, B, Nn, D);
}
